// RotaryPositionMultiHeadedAttention_27582279974924
// MI455X (gfx1250) — compile-verified, matrix operand order checked by reading, not run on hardware
//
#include <hip/hip_runtime.h>
#include <hip/hip_bf16.h>

// Problem constants (reference: B,T,C,H = 2,2048,1024,16; D = 64)
#define B_ 2
#define T_ 2048
#define C_ 1024
#define H_ 16
#define D_ 64
#define R_ (B_ * T_)            // 4096 rows for the projection GEMMs

typedef __attribute__((ext_vector_type(16))) __bf16       v16bf;
typedef __attribute__((ext_vector_type(8)))  float        v8f;
typedef __attribute__((ext_vector_type(8)))  unsigned int v8u;
typedef __attribute__((ext_vector_type(4)))  unsigned int v4u;
typedef __attribute__((ext_vector_type(8)))  int          v8i;
typedef __attribute__((ext_vector_type(4)))  int          v4i;

#define LOG2E 1.4426950408889634f

// ---- Tensor Data Mover availability (arity differs across toolchains) ----
#if defined(__has_builtin)
#if __has_builtin(__builtin_amdgcn_tensor_load_to_lds) && \
    __has_builtin(__builtin_amdgcn_s_wait_tensorcnt)
#define HAS_TDM 1
#endif
#endif

// low 32 bits of a generic LDS pointer == byte offset in LDS
__device__ __forceinline__ unsigned ldsoff(const void* p) {
  return (unsigned)(unsigned long long)p;
}

// Issue a 2-D TDM tile load (ISA ch.8 D# groups 0/1; groups 2/3 zero = <=2D).
// dims/strides in 2-byte elements; pad codes per D#: interval 2^(pi+1) dwords,
// amount (pa+1) dwords appended after each interval (reproduces LDS row pad).
__device__ __forceinline__ void tdm_load_2d(
    unsigned lds_off, const void* gaddr,
    unsigned tensor_d0, unsigned tensor_d1, unsigned long long stride_d0,
    unsigned tile_d0, unsigned tile_d1,
    unsigned pad_en, unsigned pi_code, unsigned pa_code) {
#ifdef HAS_TDM
  unsigned long long ga = (unsigned long long)gaddr;
  v4u g0;
  g0[0] = 1u;                                   // count=1, user descriptor
  g0[1] = lds_off;                              // LDS byte address
  g0[2] = (unsigned)ga;                         // global addr lo
  g0[3] = (unsigned)(ga >> 32) | (2u << 30);    // global addr hi | type=2
  v8i g1;
  unsigned cfg = (1u << 16);                    // data_size = 1 -> 2 bytes
  if (pad_en) cfg |= (1u << 20) | (pi_code << 22) | (pa_code << 25);
  g1[0] = (int)cfg;
  g1[1] = (int)((tensor_d0 & 0xFFFFu) << 16);               // dim0 lo16
  g1[2] = (int)((tensor_d0 >> 16) | ((tensor_d1 & 0xFFFFu) << 16));
  g1[3] = (int)((tensor_d1 >> 16) | (tile_d0 << 16));
  g1[4] = (int)(tile_d1 & 0xFFFFu);                         // tile_dim2 = 0
  g1[5] = (int)(unsigned)(stride_d0 & 0xFFFFFFFFull);
  g1[6] = (int)(unsigned)((stride_d0 >> 32) & 0xFFFFull);   // dim1_stride = 0
  g1[7] = 0;
  v4i z4 = {};
#if __clang_major__ >= 23
  v8i z8 = {};
  __builtin_amdgcn_tensor_load_to_lds(g0, g1, z4, z4, z8, 0);
#else
  __builtin_amdgcn_tensor_load_to_lds(g0, g1, z4, z4, 0);
#endif
#else
  (void)lds_off; (void)gaddr; (void)tensor_d0; (void)tensor_d1;
  (void)stride_d0; (void)tile_d0; (void)tile_d1;
  (void)pad_en; (void)pi_code; (void)pa_code;
#endif
}

__device__ __forceinline__ void tdm_wait() {
#ifdef HAS_TDM
  __builtin_amdgcn_s_wait_tensorcnt((short)0);
#endif
}

// ---------------- bf16 helpers (round-to-nearest-even) ----------------
__device__ __forceinline__ unsigned short f2bf(float f) {
  unsigned int u = __float_as_uint(f);
  unsigned int r = u + 0x7FFFu + ((u >> 16) & 1u);
  return (unsigned short)(r >> 16);
}

// WMMA wrapper: D = A(16x32 bf16) * B(32x16 bf16) + C(16x16 f32)
__device__ __forceinline__ v8f wmma_bf16(v8u a, v8u b, v8f c) {
  return __builtin_amdgcn_wmma_f32_16x16x32_bf16(
      false, __builtin_bit_cast(v16bf, a),
      false, __builtin_bit_cast(v16bf, b),
      (short)0, c, false, false);
}

// ISA 7.12.2 16-bit A/B fragment K-offset for vgpr v, lane-half h.
__device__ __forceinline__ int kmap(int v, int half) {
  return (v >> 2) * 16 + half * 8 + (v & 3) * 2;
}

// ---------------- Kernel 1: fused RoPE + fp32->bf16 convert ----------------
__global__ __launch_bounds__(256) void rope_prep_kernel(
    const float* __restrict__ q, const float* __restrict__ k,
    const float* __restrict__ val, const int* __restrict__ offset_p,
    unsigned short* __restrict__ qr, unsigned short* __restrict__ kr,
    unsigned short* __restrict__ vr) {
  long idx = (long)blockIdx.x * 256 + threadIdx.x;       // pair index
  const long NPAIR = (long)R_ * (C_ / 2);
  if (idx >= NPAIR) return;
  int pairInRow = (int)(idx % (C_ / 2));
  long row = idx / (C_ / 2);
  int t = (int)(row % T_);
  int d2 = pairInRow % (D_ / 2);                         // pair index in head
  int pos = offset_p[0] + t;
  float inv = __expf(-((2.0f * (float)d2) / (float)D_) * 9.2103403719761836f);
  float ang = (float)pos * inv;
  float s, c;
  __sincosf(ang, &s, &c);
  long e = idx * 2;
  float q0 = q[e], q1 = q[e + 1];
  float k0 = k[e], k1 = k[e + 1];
  qr[e]     = f2bf(q0 * c - q1 * s);
  qr[e + 1] = f2bf(q1 * c + q0 * s);
  kr[e]     = f2bf(k0 * c - k1 * s);
  kr[e + 1] = f2bf(k1 * c + k0 * s);
  vr[e]     = f2bf(val[e]);
  vr[e + 1] = f2bf(val[e + 1]);
}

// ---------------- Kernel 2: weight fp32 -> bf16, TRANSPOSED ----------------
__global__ __launch_bounds__(256) void wconv_kernel(
    const float* __restrict__ Wq, const float* __restrict__ Wk,
    const float* __restrict__ Wv, const float* __restrict__ Wo,
    unsigned short* __restrict__ wqT, unsigned short* __restrict__ wkT,
    unsigned short* __restrict__ wvT, unsigned short* __restrict__ woT) {
  int o = (blockIdx.x * 256 + threadIdx.x) * 4;
  int n = o >> 10, k0 = o & (C_ - 1);
#pragma unroll
  for (int j = 0; j < 4; ++j) {
    int k = k0 + j;
    long dst = (long)n * C_ + k;
    long src = (long)k * C_ + n;
    wqT[dst] = f2bf(Wq[src]);
    wkT[dst] = f2bf(Wk[src]);
    wvT[dst] = f2bf(Wv[src]);
    woT[dst] = f2bf(Wo[src]);
  }
}

// ---------------- Kernel 3/5: bf16 GEMM + bias (double-buffered) ----------
// mode 0: bf16 -> (B,H,T,D); mode 2: bf16 -> (B,H,D,T); mode 1: fp32 rowmajor
__global__ __launch_bounds__(256) void gemm_bias_kernel(
    const unsigned short* __restrict__ A, const unsigned short* __restrict__ WT,
    const float* __restrict__ bias, void* __restrict__ outp, int mode) {
  __shared__ unsigned short As[2][64][40];   // [buf][row][k]  (+4dw pad/row)
  __shared__ unsigned short Ws[2][64][40];   // [buf][n][k]
  const int tid = threadIdx.x;
  const int lane = tid & 31, w = tid >> 5;
  const int half = lane >> 4, ln = lane & 15;
  const int rowBase = blockIdx.x * 64;
  const int nBase = blockIdx.y * 64;
  const int wm = w & 3, wn = w >> 2;         // wave: 16 rows x 32 cols
  const int srow = tid >> 2, schunk = (tid & 3) * 8;

  // prologue: stage K-tile 0 into buffer 0
#ifdef HAS_TDM
  if (tid < 32) {  // one wave drives the DMA engine
    tdm_load_2d(ldsoff(&As[0][0][0]), &A[(long)rowBase * C_],
                C_, R_, C_, 32, 64, 1, 3, 3);   // 16dw rows + 4dw pad
    tdm_load_2d(ldsoff(&Ws[0][0][0]), &WT[(long)nBase * C_],
                C_, C_, C_, 32, 64, 1, 3, 3);
    tdm_wait();
  }
#else
  *(v4u*)&As[0][srow][schunk] =
      *(const v4u*)&A[(long)(rowBase + srow) * C_ + schunk];
  *(v4u*)&Ws[0][srow][schunk] =
      *(const v4u*)&WT[(long)(nBase + srow) * C_ + schunk];
#endif
  __syncthreads();

  v8f acc0 = {}, acc1 = {};
  for (int kb = 0; kb < C_; kb += 32) {
    int cur = (kb >> 5) & 1;
    if (kb + 32 < C_) {  // stage next tile into other buffer (overlaps WMMA)
#ifdef HAS_TDM
      if (tid < 32) {
        tdm_load_2d(ldsoff(&As[1 - cur][0][0]),
                    &A[(long)rowBase * C_ + kb + 32], C_, R_, C_, 32, 64, 1, 3, 3);
        tdm_load_2d(ldsoff(&Ws[1 - cur][0][0]),
                    &WT[(long)nBase * C_ + kb + 32], C_, C_, C_, 32, 64, 1, 3, 3);
      }
#else
      *(v4u*)&As[1 - cur][srow][schunk] =
          *(const v4u*)&A[(long)(rowBase + srow) * C_ + kb + 32 + schunk];
      *(v4u*)&Ws[1 - cur][srow][schunk] =
          *(const v4u*)&WT[(long)(nBase + srow) * C_ + kb + 32 + schunk];
#endif
    }
    v8u a, b0, b1;
#pragma unroll
    for (int v = 0; v < 8; ++v) {
      int kk = kmap(v, half);
      a[v]  = *(const unsigned int*)&As[cur][wm * 16 + ln][kk];
      b0[v] = *(const unsigned int*)&Ws[cur][wn * 32 + ln][kk];
      b1[v] = *(const unsigned int*)&Ws[cur][wn * 32 + 16 + ln][kk];
    }
    acc0 = wmma_bf16(a, b0, acc0);
    acc1 = wmma_bf16(a, b1, acc1);
#ifdef HAS_TDM
    if (tid < 32) tdm_wait();
#endif
    __syncthreads();
  }

  const int ncol0 = nBase + wn * 32 + ln;
  const int ncol1 = ncol0 + 16;
  const float bv0 = bias[ncol0], bv1 = bias[ncol1];
  if (mode == 1) {
    float* out = (float*)outp;
#pragma unroll
    for (int v = 0; v < 8; ++v) {
      int r = rowBase + wm * 16 + half * 8 + v;
      out[(long)r * C_ + ncol0] = acc0[v] + bv0;
      out[(long)r * C_ + ncol1] = acc1[v] + bv1;
    }
  } else {
    unsigned short* out = (unsigned short*)outp;
#pragma unroll
    for (int v = 0; v < 8; ++v) {
      int r = rowBase + wm * 16 + half * 8 + v;     // r = b*T + t
      int bidx = r >> 11, t = r & (T_ - 1);
      int h0 = ncol0 >> 6, d0 = ncol0 & (D_ - 1);
      int h1 = ncol1 >> 6, d1 = ncol1 & (D_ - 1);
      if (mode == 0) {  // (B,H,T,D)
        out[((long)(bidx * H_ + h0) * T_ + t) * D_ + d0] = f2bf(acc0[v] + bv0);
        out[((long)(bidx * H_ + h1) * T_ + t) * D_ + d1] = f2bf(acc1[v] + bv1);
      } else {          // mode 2: (B,H,D,T)
        out[((long)(bidx * H_ + h0) * D_ + d0) * T_ + t] = f2bf(acc0[v] + bv0);
        out[((long)(bidx * H_ + h1) * D_ + d1) * T_ + t] = f2bf(acc1[v] + bv1);
      }
    }
  }
}

// ---------------- Kernel 4: flash attention (online softmax, 2-buffered) ----
__global__ __launch_bounds__(256) void attn_kernel(
    const unsigned short* __restrict__ Qh, const unsigned short* __restrict__ Kh,
    const unsigned short* __restrict__ Vh, const unsigned char* __restrict__ mask,
    unsigned short* __restrict__ ctx) {
  __shared__ unsigned short Ks[2][32][72];     // [buf][key][d]  (+4dw pad/row)
  __shared__ unsigned short Vs[2][64][40];     // [buf][d][key]  (+4dw pad/row)
  __shared__ unsigned short Ps[8][16][36];     // per-wave P scratch
  const int tid = threadIdx.x;
  const int lane = tid & 31, w = tid >> 5;
  const int half = lane >> 4, ln = lane & 15;
  const int bh = blockIdx.y;
  const int b = bh >> 4, h = bh & 15;
  const int qrow = blockIdx.x * 128 + w * 16;
  const unsigned short* Qb = Qh + (long)bh * T_ * D_;
  const unsigned short* Kb = Kh + (long)bh * T_ * D_;
  const unsigned short* Vb = Vh + (long)bh * D_ * T_;   // (d, t) major
  const int kkey = tid >> 3, kchunk = (tid & 7) * 8;
  const int vd = tid >> 2, vchunk = (tid & 3) * 8;

  // Q fragments (16 rows x 64 d) stay in registers, A-layout.
  v8u aq0, aq1;
#pragma unroll
  for (int v = 0; v < 8; ++v) {
    int kk = kmap(v, half);
    aq0[v] = *(const unsigned int*)&Qb[(long)(qrow + ln) * D_ + kk];
    aq1[v] = *(const unsigned int*)&Qb[(long)(qrow + ln) * D_ + 32 + kk];
  }

  float rmax[8], rsum[8];
  v8f o0 = {}, o1 = {}, o2 = {}, o3 = {};
#pragma unroll
  for (int v = 0; v < 8; ++v) { rmax[v] = -1e30f; rsum[v] = 0.0f; }

  // prologue stage of key-tile 0
#ifdef HAS_TDM
  if (tid < 32) {
    tdm_load_2d(ldsoff(&Ks[0][0][0]), Kb, D_, T_, D_, 64, 32, 1, 4, 3);
    tdm_load_2d(ldsoff(&Vs[0][0][0]), Vb, T_, D_, T_, 32, 64, 1, 3, 3);
    tdm_wait();
  }
#else
  *(v4u*)&Ks[0][kkey][kchunk] = *(const v4u*)&Kb[(long)kkey * D_ + kchunk];
  *(v4u*)&Vs[0][vd][vchunk]   = *(const v4u*)&Vb[(long)vd * T_ + vchunk];
#endif
  __syncthreads();

  for (int kt = 0; kt < T_; kt += 32) {
    int cur = (kt >> 5) & 1;
    if (kt + 32 < T_) {  // stage next tile (overlaps with compute)
#ifdef HAS_TDM
      if (tid < 32) {
        tdm_load_2d(ldsoff(&Ks[1 - cur][0][0]), &Kb[(long)(kt + 32) * D_],
                    D_, T_, D_, 64, 32, 1, 4, 3);
        tdm_load_2d(ldsoff(&Vs[1 - cur][0][0]), &Vb[kt + 32],
                    T_, D_, T_, 32, 64, 1, 3, 3);
      }
#else
      *(v4u*)&Ks[1 - cur][kkey][kchunk] =
          *(const v4u*)&Kb[(long)(kt + 32 + kkey) * D_ + kchunk];
      *(v4u*)&Vs[1 - cur][vd][vchunk] =
          *(const v4u*)&Vb[(long)vd * T_ + kt + 32 + vchunk];
      if (kt + 64 < T_)
        __builtin_prefetch(&Kb[(long)(kt + 64) * D_ + tid * 8], 0, 1);
#endif
    }

    // S = Q*K^T. Tile0 = EVEN keys (2n), tile1 = ODD keys (2n+1) so each
    // lane's two probabilities are adjacent in the P spill (one b32 store).
    v8u bk;
    v8f s0 = {}, s1 = {};
#pragma unroll
    for (int v = 0; v < 8; ++v) bk[v] = *(const unsigned int*)&Ks[cur][2 * ln][kmap(v, half)];
    s0 = wmma_bf16(aq0, bk, s0);
#pragma unroll
    for (int v = 0; v < 8; ++v) bk[v] = *(const unsigned int*)&Ks[cur][2 * ln][32 + kmap(v, half)];
    s0 = wmma_bf16(aq1, bk, s0);
#pragma unroll
    for (int v = 0; v < 8; ++v) bk[v] = *(const unsigned int*)&Ks[cur][2 * ln + 1][kmap(v, half)];
    s1 = wmma_bf16(aq0, bk, s1);
#pragma unroll
    for (int v = 0; v < 8; ++v) bk[v] = *(const unsigned int*)&Ks[cur][2 * ln + 1][32 + kmap(v, half)];
    s1 = wmma_bf16(aq1, bk, s1);

    // mask + online softmax in exp2 domain
    const float SC = 0.125f * LOG2E;
    const float m0 = mask[b * T_ + kt + 2 * ln]     ? -1e30f : 0.0f;
    const float m1 = mask[b * T_ + kt + 2 * ln + 1] ? -1e30f : 0.0f;
#pragma unroll
    for (int v = 0; v < 8; ++v) {
      float x0 = s0[v] * SC + m0;
      float x1 = s1[v] * SC + m1;
      float tm = fmaxf(x0, x1);
#pragma unroll
      for (int off = 1; off < 16; off <<= 1)
        tm = fmaxf(tm, __shfl_xor(tm, off, 32));
      float nm = fmaxf(rmax[v], tm);
      float corr = exp2f(rmax[v] - nm);
      rmax[v] = nm;
      float p0 = exp2f(x0 - nm), p1 = exp2f(x1 - nm);
      float ps = p0 + p1;
#pragma unroll
      for (int off = 1; off < 16; off <<= 1)
        ps += __shfl_xor(ps, off, 32);
      rsum[v] = rsum[v] * corr + ps;
      o0[v] *= corr; o1[v] *= corr; o2[v] *= corr; o3[v] *= corr;
      *(unsigned int*)&Ps[w][half * 8 + v][2 * ln] =
          (unsigned int)f2bf(p0) | ((unsigned int)f2bf(p1) << 16);
    }

    // O += P(16x32) * V(32x64)   (DS ops are in-order within the wave)
    v8u pa, bvv;
#pragma unroll
    for (int v = 0; v < 8; ++v) pa[v] = *(const unsigned int*)&Ps[w][ln][kmap(v, half)];
#pragma unroll
    for (int v = 0; v < 8; ++v) bvv[v] = *(const unsigned int*)&Vs[cur][ln][kmap(v, half)];
    o0 = wmma_bf16(pa, bvv, o0);
#pragma unroll
    for (int v = 0; v < 8; ++v) bvv[v] = *(const unsigned int*)&Vs[cur][16 + ln][kmap(v, half)];
    o1 = wmma_bf16(pa, bvv, o1);
#pragma unroll
    for (int v = 0; v < 8; ++v) bvv[v] = *(const unsigned int*)&Vs[cur][32 + ln][kmap(v, half)];
    o2 = wmma_bf16(pa, bvv, o2);
#pragma unroll
    for (int v = 0; v < 8; ++v) bvv[v] = *(const unsigned int*)&Vs[cur][48 + ln][kmap(v, half)];
    o3 = wmma_bf16(pa, bvv, o3);
#ifdef HAS_TDM
    if (tid < 32) tdm_wait();
#endif
    __syncthreads();
  }

  // epilogue: normalize and write ctx in (B,T,H*D) bf16 row-major
#pragma unroll
  for (int v = 0; v < 8; ++v) {
    float inv = 1.0f / rsum[v];
    int t = qrow + half * 8 + v;
    long base = ((long)(b * T_ + t)) * C_ + h * D_;
    ctx[base + 0 * 16 + ln] = f2bf(o0[v] * inv);
    ctx[base + 1 * 16 + ln] = f2bf(o1[v] * inv);
    ctx[base + 2 * 16 + ln] = f2bf(o2[v] * inv);
    ctx[base + 3 * 16 + ln] = f2bf(o3[v] * inv);
  }
}

// ---------------- launcher ----------------
extern "C" void kernel_launch(void* const* d_in, const int* in_sizes, int n_in,
                              void* d_out, int out_size, void* d_ws, size_t ws_size,
                              hipStream_t stream) {
  const float* q    = (const float*)d_in[0];
  const float* k    = (const float*)d_in[1];
  const float* v    = (const float*)d_in[2];
  const unsigned char* mask = (const unsigned char*)d_in[3];  // jnp bool -> 1B
  const int*   off  = (const int*)d_in[4];
  const float* Wq   = (const float*)d_in[5];
  const float* bq   = (const float*)d_in[6];
  const float* Wk   = (const float*)d_in[7];
  const float* bk   = (const float*)d_in[8];
  const float* Wv   = (const float*)d_in[9];
  const float* bv   = (const float*)d_in[10];
  const float* Wo   = (const float*)d_in[11];
  const float* bo   = (const float*)d_in[12];

  const size_t MB = 1024ull * 1024ull;
  unsigned char* ws = (unsigned char*)d_ws;
  unsigned short* qr   = (unsigned short*)(ws + 0 * MB);   // roped q (reused as ctx)
  unsigned short* kr   = (unsigned short*)(ws + 8 * MB);
  unsigned short* vr   = (unsigned short*)(ws + 16 * MB);
  unsigned short* Qh   = (unsigned short*)(ws + 24 * MB);  // (B,H,T,D)
  unsigned short* Kh   = (unsigned short*)(ws + 32 * MB);  // (B,H,T,D)
  unsigned short* Vh   = (unsigned short*)(ws + 40 * MB);  // (B,H,D,T)
  unsigned short* wqT  = (unsigned short*)(ws + 48 * MB);  // transposed weights
  unsigned short* wkT  = (unsigned short*)(ws + 50 * MB);
  unsigned short* wvT  = (unsigned short*)(ws + 52 * MB);
  unsigned short* woT  = (unsigned short*)(ws + 54 * MB);
  unsigned short* ctx  = qr;

  rope_prep_kernel<<<(R_ * (C_ / 2)) / 256, 256, 0, stream>>>(q, k, v, off, qr, kr, vr);
  wconv_kernel<<<(C_ * C_) / (256 * 4), 256, 0, stream>>>(Wq, Wk, Wv, Wo, wqT, wkT, wvT, woT);
  dim3 gg(R_ / 64, C_ / 64);
  gemm_bias_kernel<<<gg, 256, 0, stream>>>(qr, wqT, bq, Qh, 0);
  gemm_bias_kernel<<<gg, 256, 0, stream>>>(kr, wkT, bk, Kh, 0);
  gemm_bias_kernel<<<gg, 256, 0, stream>>>(vr, wvT, bv, Vh, 2);
  attn_kernel<<<dim3(T_ / 128, B_ * H_), 256, 0, stream>>>(Qh, Kh, Vh, mask, ctx);
  gemm_bias_kernel<<<gg, 256, 0, stream>>>(ctx, woT, bo, d_out, 1);
}
